// DynamicKernelSpatialAttention_41609643163803
// MI455X (gfx1250) — compile-verified, matrix operand order checked by reading, not run on hardware
//
#include <hip/hip_runtime.h>
#include <hip/hip_bf16.h>

typedef __attribute__((ext_vector_type(16))) _Float16 v16h;
typedef __attribute__((ext_vector_type(8)))  _Float16 v8h;
typedef __attribute__((ext_vector_type(8)))  float    v8f;

#define Bx 32
#define Cc 64
#define Hh_ 256
#define Ww 256
#define HW (Hh_ * Ww)

// ---------------------------------------------------------------------------
// Kernel 1: one pass over x computes BOTH reductions, b128 (float4) loads.
//   x_mean[b,h,w] = mean_c x[b,c,h,w]
//   pooled[b,c]   = sum_{h,w} x[b,c,h,w]
// Block = 4 rows of one sample: thread (row, w4) loads float4 per channel.
// 512 B per wave per vmem instruction -> HBM-saturating access width.
// ---------------------------------------------------------------------------
__global__ __launch_bounds__(256) void fused_reduce_kernel(
    const float* __restrict__ x, float* __restrict__ pooled_sum,
    float* __restrict__ xmean) {
  const int blk = blockIdx.x;       // b*(H/4) + hq
  const int b = blk >> 6;
  const int hq = blk & 63;
  const int row = threadIdx.x >> 6; // 0..3
  const int w4 = threadIdx.x & 63;  // float4 index within row
  const int h = hq * 4 + row;

  __shared__ float lds_pool[Cc];
  if (threadIdx.x < Cc) lds_pool[threadIdx.x] = 0.0f;
  __syncthreads();

  const float* base =
      x + ((size_t)b * Cc) * HW + (size_t)h * Ww + (size_t)w4 * 4;
  float4 csum = make_float4(0.f, 0.f, 0.f, 0.f);
#pragma unroll 4
  for (int c = 0; c < Cc; ++c) {
    const float4 v = *(const float4*)(base + (size_t)c * HW);
    csum.x += v.x;
    csum.y += v.y;
    csum.z += v.z;
    csum.w += v.w;
    // per-lane horizontal sum, then wave32 butterfly reduction
    float r = (v.x + v.y) + (v.z + v.w);
    r += __shfl_xor(r, 16, 32);
    r += __shfl_xor(r, 8, 32);
    r += __shfl_xor(r, 4, 32);
    r += __shfl_xor(r, 2, 32);
    r += __shfl_xor(r, 1, 32);
    if ((threadIdx.x & 31) == 0) atomicAdd(&lds_pool[c], r);
  }
  const float sc = 1.0f / (float)Cc;
  float4 m = make_float4(csum.x * sc, csum.y * sc, csum.z * sc, csum.w * sc);
  *(float4*)(xmean + (size_t)b * HW + (size_t)h * Ww + (size_t)w4 * 4) = m;
  __syncthreads();
  if (threadIdx.x < Cc)
    atomicAdd(&pooled_sum[b * Cc + threadIdx.x], lds_pool[threadIdx.x]);
}

// ---------------------------------------------------------------------------
// Kernel 2: kernel-generator MLP on tensor cores (v_wmma_f32_16x16x32_f16).
//   h  = relu(pooled_mean @ w1^T + b1)    M=32 N=64 K=64
//   kg = h @ w2^T + b2                    M=32 N=16(9 padded) K=64
// ---------------------------------------------------------------------------
__device__ inline v16h load_frag(const _Float16* tile, int lane, int kk) {
  const int row = lane & 15;
  const int s = (lane < 16) ? 0 : 8;
  const _Float16* rp = tile + row * 64 + kk * 32 + s;
  union {
    v16h v;
    v8h h[2];
  } f;
  f.h[0] = *(const v8h*)(rp);
  f.h[1] = *(const v8h*)(rp + 16);
  return f.v;
}

__global__ __launch_bounds__(256) void kgen_wmma_kernel(
    const float* __restrict__ pooled_sum, const float* __restrict__ w1,
    const float* __restrict__ b1, const float* __restrict__ w2,
    const float* __restrict__ b2, float* __restrict__ kern) {
  __shared__ __align__(16) _Float16 Ah[Bx][64];    // pooled mean, f16
  __shared__ __align__(16) _Float16 Bh[64][64];    // w1 rows = N, cols = K
  __shared__ __align__(16) _Float16 Hd[Bx][64];    // hidden activations
  __shared__ __align__(16) _Float16 W2h[16][64];   // w2 padded 9->16 rows

  const int t = threadIdx.x;
  for (int i = t; i < Bx * 64; i += 256)
    Ah[0][i] = (_Float16)(pooled_sum[i] * (1.0f / (float)HW));
  for (int i = t; i < 64 * 64; i += 256) Bh[0][i] = (_Float16)w1[i];
  for (int i = t; i < 16 * 64; i += 256) {
    int r = i >> 6, c = i & 63;
    W2h[0][i] = (_Float16)((r < 9) ? w2[r * 64 + c] : 0.0f);
  }
  __syncthreads();

  const int wave = t >> 5;
  const int lane = t & 31;
  const int n = lane & 15;
  const int mbase = (lane < 16) ? 0 : 8;

  // Stage A: 2 M-tiles x 4 N-tiles across 8 waves, K=64 in 2 steps
  {
    const int tm = wave >> 2;
    const int tn = wave & 3;
    v8f acc = {};
#pragma unroll
    for (int kk = 0; kk < 2; ++kk) {
      v16h a = load_frag(&Ah[tm * 16][0], lane, kk);
      v16h bm = load_frag(&Bh[tn * 16][0], lane, kk);
      acc = __builtin_amdgcn_wmma_f32_16x16x32_f16(false, a, false, bm,
                                                   (short)0, acc, false, false);
    }
    const float bias = b1[tn * 16 + n];
#pragma unroll
    for (int v = 0; v < 8; ++v) {
      float val = acc[v] + bias;
      val = val > 0.0f ? val : 0.0f;  // relu
      Hd[tm * 16 + mbase + v][tn * 16 + n] = (_Float16)val;
    }
  }
  __syncthreads();

  // Stage B: 2 M-tiles, single N tile (9 outputs padded to 16), K=64
  if (wave < 2) {
    v8f acc = {};
#pragma unroll
    for (int kk = 0; kk < 2; ++kk) {
      v16h a = load_frag(&Hd[wave * 16][0], lane, kk);
      v16h bm = load_frag(&W2h[0][0], lane, kk);
      acc = __builtin_amdgcn_wmma_f32_16x16x32_f16(false, a, false, bm,
                                                   (short)0, acc, false, false);
    }
    if (n < 9) {
      const float bias = b2[n];
#pragma unroll
      for (int v = 0; v < 8; ++v) {
        int m = wave * 16 + mbase + v;  // batch index
        kern[m * 9 + n] = acc[v] + bias;
      }
    }
  }
}

// ---------------------------------------------------------------------------
// Kernel 3: fused dynamic 3x3 cross-correlation + sigmoid + apply.
// Block = 4 rows of one sample; each thread computes 4 att values from a
// 3x6 x_mean window (cached in L2/WGP$), then streams all 64 channels with
// b128 load/store pairs.
// ---------------------------------------------------------------------------
__global__ __launch_bounds__(256) void apply_att_kernel(
    const float* __restrict__ x, const float* __restrict__ xmean,
    const float* __restrict__ kern, float* __restrict__ out) {
  const int blk = blockIdx.x;       // b*(H/4) + hq
  const int b = blk >> 6;
  const int hq = blk & 63;
  const int row = threadIdx.x >> 6; // 0..3
  const int w4 = threadIdx.x & 63;
  const int h = hq * 4 + row;

  __shared__ float kk[9];
  if (threadIdx.x < 9) kk[threadIdx.x] = kern[b * 9 + threadIdx.x];
  __syncthreads();

  // Build 3x6 window of x_mean around the 4 output columns [w4*4 .. w4*4+3]
  float win[3][6];
#pragma unroll
  for (int r = 0; r < 3; ++r) {
    const int hy = h + r - 1;
    if (hy < 0 || hy > Hh_ - 1) {
#pragma unroll
      for (int j = 0; j < 6; ++j) win[r][j] = 0.0f;
    } else {
      const float* mr = xmean + (size_t)b * HW + (size_t)hy * Ww + w4 * 4;
      const float4 cv = *(const float4*)mr;
      win[r][0] = (w4 > 0) ? mr[-1] : 0.0f;
      win[r][1] = cv.x;
      win[r][2] = cv.y;
      win[r][3] = cv.z;
      win[r][4] = cv.w;
      win[r][5] = (w4 < 63) ? mr[4] : 0.0f;
    }
  }

  float4 att;
  float* attp = &att.x;
#pragma unroll
  for (int j = 0; j < 4; ++j) {
    float s = 0.0f;
#pragma unroll
    for (int r = 0; r < 3; ++r)
#pragma unroll
      for (int dx = 0; dx < 3; ++dx) s += kk[r * 3 + dx] * win[r][j + dx];
    attp[j] = 1.0f / (1.0f + __expf(-s));
  }

  const size_t base =
      ((size_t)b * Cc) * HW + (size_t)h * Ww + (size_t)w4 * 4;
#pragma unroll 4
  for (int c = 0; c < Cc; ++c) {
    const float4 v = *(const float4*)(x + base + (size_t)c * HW);
    float4 o;
    o.x = v.x * att.x;
    o.y = v.y * att.y;
    o.z = v.z * att.z;
    o.w = v.w * att.w;
    *(float4*)(out + base + (size_t)c * HW) = o;
  }
}

extern "C" void kernel_launch(void* const* d_in, const int* in_sizes, int n_in,
                              void* d_out, int out_size, void* d_ws,
                              size_t ws_size, hipStream_t stream) {
  (void)in_sizes;
  (void)n_in;
  (void)out_size;
  (void)ws_size;
  const float* x = (const float*)d_in[0];
  const float* w1 = (const float*)d_in[1];
  const float* b1 = (const float*)d_in[2];
  const float* w2 = (const float*)d_in[3];
  const float* b2 = (const float*)d_in[4];
  float* out = (float*)d_out;

  // workspace layout (floats): pooled[2048] | kern[288->pad 512] | xmean[32*65536]
  float* pooled = (float*)d_ws;
  float* kern = pooled + 2048;
  float* xmean = pooled + 2560;  // 10240-byte offset: 16B aligned for float4

  hipMemsetAsync(pooled, 0, 2048 * sizeof(float), stream);

  fused_reduce_kernel<<<Bx * (Hh_ / 4), 256, 0, stream>>>(x, pooled, xmean);
  kgen_wmma_kernel<<<1, 256, 0, stream>>>(pooled, w1, b1, w2, b2, kern);
  apply_att_kernel<<<Bx * (Hh_ / 4), 256, 0, stream>>>(x, xmean, kern, out);
}
